// Encoder_15857019257279
// MI455X (gfx1250) — matrix operand order repaired and checked by reading, not compiled, not run on hardware
//
#include <hip/hip_runtime.h>
#include <hip/hip_bf16.h>
#include <stdint.h>

// Problem constants (from reference)
#define S_LEN 256
#define BATCH 64
#define EMB   512
#define HID   1024
#define G4H   (4*HID)

typedef __attribute__((ext_vector_type(16))) _Float16 v16h;
typedef __attribute__((ext_vector_type(8)))  _Float16 v8h;
typedef __attribute__((ext_vector_type(8)))  float    v8f;
typedef __attribute__((ext_vector_type(4)))  int      v4i;

typedef __attribute__((address_space(1))) v4i* gptr_v4i;
typedef __attribute__((address_space(3))) v4i* lptr_v4i;

__device__ __forceinline__ v16h cat8(v8h lo, v8h hi) {
  return __builtin_shufflevector(lo, hi, 0,1,2,3,4,5,6,7,8,9,10,11,12,13,14,15);
}
// A-operand (16x32 f16): lane half lkg holds K = [lkg*8,+8) and [16+lkg*8,+8); ptr includes +lkg*8
__device__ __forceinline__ v16h load_a(const _Float16* __restrict__ p) {
  return cat8(*(const v8h*)p, *(const v8h*)(p + 16));
}
// B-operand (32x16 f16): lane = column n, lane half lkg holds K = [lkg*8,+8) and [16+lkg*8,+8); ptr includes +lkg*8
__device__ __forceinline__ v16h load_b(const _Float16* __restrict__ p) {
  return cat8(*(const v8h*)p, *(const v8h*)(p + 16));
}

__device__ __forceinline__ float fsig(float x) {
  return __builtin_amdgcn_rcpf(1.0f + __builtin_amdgcn_exp2f(-1.442695041f * x));
}
__device__ __forceinline__ float ftanh(float x) {
  return 2.0f * __builtin_amdgcn_rcpf(1.0f + __builtin_amdgcn_exp2f(-2.885390082f * x)) - 1.0f;
}

#define WMMA_F16(A, Bm, C) \
  __builtin_amdgcn_wmma_f32_16x16x32_f16(false, (A), false, (Bm), (short)0, (C), false, false)

#define SPLAT8(x) {(x),(x),(x),(x),(x),(x),(x),(x)}

// ---- CDNA5 async global->LDS copy (ASYNCcnt-tracked), builtin with asm fallback ----
#if defined(__has_builtin) && __has_builtin(__builtin_amdgcn_global_load_async_to_lds_b128)
#define ASYNC_LD_B128(gp, lp)                                       \
  __builtin_amdgcn_global_load_async_to_lds_b128(                   \
      (gptr_v4i)(uintptr_t)(gp),                                    \
      (lptr_v4i)(uintptr_t)(unsigned)(uintptr_t)(lp),               \
      0, 0)
#else
#define ASYNC_LD_B128(gp, lp)                                       \
  asm volatile("global_load_async_to_lds_b128 %0, %1, off"          \
               :: "v"((unsigned)(uintptr_t)(lp)), "v"((const void*)(gp)) : "memory")
#endif

#if defined(__has_builtin) && __has_builtin(__builtin_amdgcn_s_wait_asynccnt)
#define WAIT_ASYNC(n) __builtin_amdgcn_s_wait_asynccnt(n)
#else
#define WAIT_ASYNC(n) asm volatile("s_wait_asynccnt %0" :: "i"(n) : "memory")
#endif

// ---------------------------------------------------------------------------
// Prep kernels
// ---------------------------------------------------------------------------
__global__ void cvt_f16_kernel(const float* __restrict__ in, _Float16* __restrict__ out, int n) {
  int i = (blockIdx.x * blockDim.x + threadIdx.x) * 4;
  if (i < n) {
#pragma unroll
    for (int j = 0; j < 4; ++j) out[i + j] = (_Float16)in[i + j];
  }
}

__global__ void bias_sum_kernel(const float* __restrict__ a, const float* __restrict__ b,
                                float* __restrict__ o, int n) {
  int i = blockIdx.x * blockDim.x + threadIdx.x;
  if (i < n) o[i] = a[i] + b[i];
}

__global__ void zero_kernel(float* __restrict__ p, int n) {
  int i = blockIdx.x * blockDim.x + threadIdx.x;
  if (i < n) p[i] = 0.0f;
}

// Embedding gather + fp32->f16 convert: out[s,b,e] = (f16)table[src[s,b], e]
__global__ void gather_kernel(const int* __restrict__ src, const float* __restrict__ tab,
                              _Float16* __restrict__ out) {
  size_t i  = (size_t)blockIdx.x * blockDim.x + threadIdx.x;  // over S*B*(E/8)
  size_t sb = i >> 6;                 // E/8 == 64
  int    e  = (int)(i & 63) << 3;
  int    id = src[sb];
  const float* p = tab + (size_t)id * EMB + e;
  v8h o;
#pragma unroll
  for (int j = 0; j < 8; ++j) o[j] = (_Float16)p[j];
  *(v8h*)(out + sb * EMB + e) = o;
}

// ---------------------------------------------------------------------------
// One recurrent step, both directions fused.
// Workgroup = 4 waves = the 4 batch tiles of one (dir, hidden-tile).
// B operand (4 gate tiles per 32-K slice, 4KB) is staged once per WG into LDS
// via double-buffered GLOBAL_LOAD_ASYNC_TO_LDS_B128; waves read it with
// ds_load_b128. A operand stays as direct global_load_b128 (per-wave private).
// grid = 128 blocks (2 dir x 64 htile) x 128 threads.
// ---------------------------------------------------------------------------
__global__ __launch_bounds__(128) void lstm_step_kernel(
    const _Float16* __restrict__ embed_h,  // [S,B,E] f16
    const _Float16* __restrict__ wih_h,    // [2][4H,E] f16
    const _Float16* __restrict__ whh_h,    // [2][4H,H] f16
    const float*    __restrict__ bias,     // [2][4H]  (b_ih + b_hh)
    _Float16*       __restrict__ hbuf,     // [2][2][B,H] f16 ping-pong
    float*          __restrict__ cbuf,     // [2][B,H] fp32
    float*          __restrict__ out,      // d_out: hs_f|cs_f|hs_b|cs_b|hs
    int t)
{
  __shared__ _Float16 sB[2][4][16][32];   // [buf][gate][row(n)][k]  = 8KB

  const int tid  = threadIdx.x;
  const int lane = tid & 31;
  const int mb   = tid >> 5;                 // batch tile = wave id in WG
  const int dir  = blockIdx.x >> 6;          // 0 fwd, 1 bwd
  const int nh   = blockIdx.x & 63;          // hidden tile
  const int tIdx = dir ? (S_LEN - 1 - t) : t;
  const int lrow = lane & 15;
  const int lkg  = lane >> 4;
  const int brow = (mb << 4) + lrow;         // batch row (A, M)
  const int hcol = (nh << 4) + lrow;         // hidden col (B/C, N)

  // A operands
  const _Float16* aE = embed_h + ((size_t)(tIdx * BATCH + brow)) * EMB + (lkg << 3);
  const _Float16* aH = hbuf + ((size_t)((dir * 2 + (t & 1)) * BATCH + brow)) * HID + (lkg << 3);

  // weight bases for this direction
  const _Float16* wihD = wih_h + (size_t)dir * G4H * EMB;
  const _Float16* whhD = whh_h + (size_t)dir * G4H * HID;

  // stage one 32-K slice of all 4 gate tiles (4KB) into sB[buf]:
  // 256 x 16B chunks; each thread moves chunks tid and tid+128.
  auto stage = [&](int s, int buf) {
#pragma unroll
    for (int c2 = 0; c2 < 2; ++c2) {
      int c   = tid + (c2 << 7);          // 0..255
      int g   = c >> 6;                   // gate
      int row = (c >> 2) & 15;            // n within tile
      int k16 = c & 3;                    // 16B chunk within 64B row-slice
      const _Float16* gp;
      if (s < 16)
        gp = wihD + ((size_t)(g * HID + (nh << 4) + row)) * EMB + (s << 5) + (k16 << 3);
      else
        gp = whhD + ((size_t)(g * HID + (nh << 4) + row)) * HID + ((s - 16) << 5) + (k16 << 3);
      ASYNC_LD_B128(gp, &sB[buf][g][row][k16 << 3]);
    }
  };

  // bias -> fp32 accumulators (bias depends only on N = lane column)
  const float* bsD = bias + dir * G4H;
  const float  bv0 = bsD[0 * HID + hcol];
  const float  bv1 = bsD[1 * HID + hcol];
  const float  bv2 = bsD[2 * HID + hcol];
  const float  bv3 = bsD[3 * HID + hcol];
  v8f acc0 = SPLAT8(bv0);
  v8f acc1 = SPLAT8(bv1);
  v8f acc2 = SPLAT8(bv2);
  v8f acc3 = SPLAT8(bv3);

  // prime the pipeline with slice 0
  stage(0, 0);

  // 48 K-slices: 16 over E (input projection) + 32 over H (recurrence)
  for (int s = 0; s < 48; ++s) {
    if (s > 0) __syncthreads();            // protect buffer being restaged below
    if (s + 1 < 48) {
      stage(s + 1, (s + 1) & 1);
      WAIT_ASYNC(2);                       // in-order: slice s's 2 copies have landed
    } else {
      WAIT_ASYNC(0);
    }
    __syncthreads();                       // staged data visible to all 4 waves

    const _Float16* ap = (s < 16) ? (aE + (s << 5)) : (aH + ((s - 16) << 5));
    v16h a = load_a(ap);

    const _Float16* lb = &sB[s & 1][0][lrow][lkg << 3];
    acc0 = WMMA_F16(a, load_b(lb + 0 * 512), acc0);   // gate stride = 16*32 halves
    acc1 = WMMA_F16(a, load_b(lb + 1 * 512), acc1);
    acc2 = WMMA_F16(a, load_b(lb + 2 * 512), acc2);
    acc3 = WMMA_F16(a, load_b(lb + 3 * 512), acc3);
  }

  // ---- fused gates in accumulator layout ----
  // C/D layout: lanes 0-15 -> N=lane, M=vgpr j; lanes 16-31 -> N=lane-16, M=j+8
  const size_t SBH = (size_t)S_LEN * BATCH * HID;
  float* hs_out = out + (size_t)dir * 2 * SBH + (size_t)tIdx * BATCH * HID;
  float* cs_out = hs_out + SBH;
  float* cD     = cbuf + (size_t)dir * BATCH * HID;
  _Float16* hN  = hbuf + ((size_t)((dir * 2 + ((t + 1) & 1)) * BATCH)) * HID;

  const int m0 = lkg << 3;
#pragma unroll
  for (int j = 0; j < 8; ++j) {
    size_t idx = (size_t)((mb << 4) + m0 + j) * HID + hcol;
    float iv = fsig(acc0[j]);
    float fv = fsig(acc1[j]);
    float gv = ftanh(acc2[j]);
    float ov = fsig(acc3[j]);
    float cp = (t == 0) ? 0.0f : cD[idx];
    float cn = fv * cp + iv * gv;
    float hn = ov * ftanh(cn);
    cD[idx]     = cn;
    cs_out[idx] = cn;
    hs_out[idx] = hn;
    hN[idx]     = (_Float16)hn;   // next step's A operand
  }
}

// hs = hs_f[S-1] + hs_b[S-1]  (both time-aligned at last index)
__global__ void final_add_kernel(float* __restrict__ out) {
  int i = blockIdx.x * blockDim.x + threadIdx.x;          // over B*H
  const size_t SBH  = (size_t)S_LEN * BATCH * HID;
  const size_t last = (size_t)(S_LEN - 1) * BATCH * HID;
  out[4 * SBH + i] = out[last + i] + out[2 * SBH + last + i];
}

// ---------------------------------------------------------------------------
extern "C" void kernel_launch(void* const* d_in, const int* in_sizes, int n_in,
                              void* d_out, int out_size, void* d_ws, size_t ws_size,
                              hipStream_t stream) {
  (void)in_sizes; (void)n_in; (void)out_size; (void)ws_size;

  const int*   source = (const int*)  d_in[0];
  const float* table  = (const float*)d_in[1];
  const float* Wih[2] = {(const float*)d_in[2], (const float*)d_in[6]};
  const float* Whh[2] = {(const float*)d_in[3], (const float*)d_in[7]};
  const float* bih[2] = {(const float*)d_in[4], (const float*)d_in[8]};
  const float* bhh[2] = {(const float*)d_in[5], (const float*)d_in[9]};
  float* out = (float*)d_out;

  // ---- workspace carve (~41 MB total) ----
  constexpr size_t WIH_ELEMS = (size_t)G4H * EMB;   // 2M / dir
  constexpr size_t WHH_ELEMS = (size_t)G4H * HID;   // 4M / dir
  constexpr size_t OFF_WIH  = 0;
  constexpr size_t OFF_WHH  = OFF_WIH  + 2 * WIH_ELEMS * 2;
  constexpr size_t OFF_BIAS = OFF_WHH  + 2 * WHH_ELEMS * 2;
  constexpr size_t OFF_EMB  = OFF_BIAS + 2 * (size_t)G4H * 4;
  constexpr size_t OFF_HBUF = OFF_EMB  + (size_t)S_LEN * BATCH * EMB * 2;
  constexpr size_t OFF_CBUF = OFF_HBUF + 2 * 2 * (size_t)BATCH * HID * 2;

  char* ws = (char*)d_ws;
  _Float16* wih_h = (_Float16*)(ws + OFF_WIH);
  _Float16* whh_h = (_Float16*)(ws + OFF_WHH);
  float*    bias  = (float*)   (ws + OFF_BIAS);
  _Float16* emb_h = (_Float16*)(ws + OFF_EMB);
  _Float16* hbuf  = (_Float16*)(ws + OFF_HBUF);
  float*    cbuf  = (float*)   (ws + OFF_CBUF);

  // prep: convert weights to f16 (stay resident in L2), bias sums, h0 = 0
  for (int d = 0; d < 2; ++d) {
    cvt_f16_kernel<<<(int)(WIH_ELEMS / 4 / 256), 256, 0, stream>>>(
        Wih[d], wih_h + d * WIH_ELEMS, (int)WIH_ELEMS);
    cvt_f16_kernel<<<(int)(WHH_ELEMS / 4 / 256), 256, 0, stream>>>(
        Whh[d], whh_h + d * WHH_ELEMS, (int)WHH_ELEMS);
    bias_sum_kernel<<<G4H / 256, 256, 0, stream>>>(bih[d], bhh[d], bias + d * G4H, G4H);
  }
  zero_kernel<<<(2 * 2 * BATCH * HID * 2 / 4) / 256, 256, 0, stream>>>(
      (float*)hbuf, 2 * 2 * BATCH * HID / 2);
  // embedding gather -> f16
  gather_kernel<<<(S_LEN * BATCH * (EMB / 8)) / 256, 256, 0, stream>>>(source, table, emb_h);

  // sequential recurrence: forward step t and backward step S-1-t in one launch
  for (int t = 0; t < S_LEN; ++t) {
    lstm_step_kernel<<<128, 128, 0, stream>>>(emb_h, wih_h, whh_h, bias, hbuf, cbuf, out, t);
  }

  final_add_kernel<<<(BATCH * HID) / 256, 256, 0, stream>>>(out);
}
